// GraphTransformerNetwork_31963146617558
// MI455X (gfx1250) — matrix operand order repaired and checked by reading, not compiled, not run on hardware
//
#include <hip/hip_runtime.h>
#include <hip/hip_bf16.h>
#include <math.h>

#define N_NODES 100000
#define N_EDGES 800000
#define IN_DIM  128
#define HID     32
#define HEADS   4
#define QKV_DIM (HEADS * HID)   // 128
#define OUT_DIM 16

typedef __attribute__((ext_vector_type(16))) _Float16 v16h;
typedef __attribute__((ext_vector_type(8)))  float    v8f;
typedef __attribute__((ext_vector_type(2)))  _Float16 h2;

// ---------------- utility kernels ----------------

__global__ void fill_f32(float* p, float v, int n) {
    int i = blockIdx.x * blockDim.x + threadIdx.x;
    if (i < n) p[i] = v;
}

__global__ void cvt_f32_f16(const float* __restrict__ src, _Float16* __restrict__ dst, int n) {
    int i = blockIdx.x * blockDim.x + threadIdx.x;
    if (i < n) dst[i] = (_Float16)src[i];
}

// W [D x F] f32 row-major  ->  WT [F x D] f16 (K-contiguous per output column)
__global__ void wt_prep(const float* __restrict__ W, _Float16* __restrict__ WT, int D, int F) {
    int i = blockIdx.x * blockDim.x + threadIdx.x;
    if (i >= D * F) return;
    int d = i / F, f = i - d * F;
    WT[f * D + d] = (_Float16)W[i];
}

// ---------------- WMMA GEMM ----------------
// C[M x F] = X[M x D] @ W[D x F] + bias, X f16 row-major, WT = W^T f16 [F x D].
// The whole weight matrix (<=32 KB f16) is staged into LDS once per block and
// shared by all 4 waves, so inner-loop B fetches are low-latency ds_load_b128
// (DScnt) instead of L2 round-trips, decoupled from the A-stream (LOADcnt).
// One wave owns a 16-row m-tile and ALL NT = F/16 n-tiles: the A fragment is
// loaded once per k-step and reused across NT back-to-back WMMAs, so X is
// streamed exactly once per GEMM.
// M multiple of 16, D multiple of 32 (true for all call sites).
template <int D, int NT, int OUT_F16>
__global__ void __launch_bounds__(128)
wmma_gemm(const _Float16* __restrict__ X, const _Float16* __restrict__ WT,
          const float* __restrict__ bias, void* __restrict__ Cout, int M) {
    constexpr int F = NT * 16;

    // ---- stage WT into LDS (whole block cooperates, before any wave exits) ----
    __shared__ float4 wlds4[(F * D) / 8];          // F*D f16 = F*D*2 bytes
    {
        const float4* gsrc = (const float4*)WT;
#pragma unroll 2
        for (int i = threadIdx.x; i < (F * D) / 8; i += 128)
            wlds4[i] = gsrc[i];
    }
    __syncthreads();
    const _Float16* wlds = (const _Float16*)wlds4;

    int wave = blockIdx.x * (blockDim.x >> 5) + (threadIdx.x >> 5);
    int lane = threadIdx.x & 31;
    if (wave >= (M >> 4)) return;              // wave-uniform: EXEC stays all-ones

    int tm   = wave << 4;
    int half = lane >> 4;                      // 0: lanes 0-15, 1: lanes 16-31
    int l16  = lane & 15;

    const _Float16* xrow = X + (size_t)(tm + l16) * D;   // A: row M = l16

    v8f zero = {};
    v8f acc[NT];
#pragma unroll
    for (int t = 0; t < NT; ++t) acc[t] = zero;

#pragma unroll
    for (int k0 = 0; k0 < D; k0 += 32) {
        // A k order here (contiguous): VGPR j<4 -> K = k0 + 16*half + 2j(+1)
        //                     VGPR j>=4 -> K = k0 + 16*half + 8 + 2(j-4)(+1)
        v16h a;
        const h2* xa = (const h2*)(xrow + k0 + half * 16);
        const h2* xb = (const h2*)(xrow + k0 + half * 16 + 8);
#pragma unroll
        for (int j = 0; j < 4; ++j) {
            h2 t0 = xa[j];  a[2*j]     = t0.x;  a[2*j + 1]     = t0.y;
            h2 t1 = xb[j];  a[8 + 2*j] = t1.x;  a[8 + 2*j + 1] = t1.y;
        }

#pragma unroll
        for (int t = 0; t < NT; ++t) {
            // B k order here (contiguous): lanes 0-15 hold K=k0..k0+15, lanes 16-31 K=k0+16..k0+31
            // 16 consecutive f16 per lane from LDS -> two ds_load_b128
            const h2* wb = (const h2*)(wlds + (t * 16 + l16) * D + half * 16 + k0);
            v16h b;
#pragma unroll
            for (int j = 0; j < 8; ++j) {
                h2 w = wb[j];  b[2*j] = w.x;  b[2*j + 1] = w.y;
            }
            acc[t] = __builtin_amdgcn_wmma_f32_16x16x32_f16(
                         false, a, false, b, (short)0, acc[t], false, false);
        }
    }

    // C/D layout: VGPR i -> row (i + 8*half), col l16
#pragma unroll
    for (int t = 0; t < NT; ++t) {
        float bv = bias[t * 16 + l16];
        if (OUT_F16) {
            _Float16* C = (_Float16*)Cout;
#pragma unroll
            for (int i = 0; i < 8; ++i)
                C[(size_t)(tm + i + half * 8) * F + t * 16 + l16] = (_Float16)(acc[t][i] + bv);
        } else {
            float* C = (float*)Cout;
#pragma unroll
            for (int i = 0; i < 8; ++i)
                C[(size_t)(tm + i + half * 8) * F + t * 16 + l16] = acc[t][i] + bv;
        }
    }
}

// ---------------- edge passes ----------------

__device__ __forceinline__ void atomicMaxF(float* addr, float val) {
    if (val >= 0.0f) atomicMax((int*)addr, __float_as_int(val));
    else             atomicMin((unsigned int*)addr, (unsigned int)__float_as_int(val));
}

// alpha[e,h] = scale * <q[dst], k[src]> ; segment-max into mmax[dst,h]
__global__ void edge_pass_a(const _Float16* __restrict__ q, const _Float16* __restrict__ k,
                            const int* __restrict__ src, const int* __restrict__ dst,
                            float* __restrict__ alpha, float* __restrict__ mmax, int E) {
    int idx = blockIdx.x * blockDim.x + threadIdx.x;
    if (idx >= E * HEADS) return;
    int e = idx >> 2, h = idx & 3;
    int s = src[e], d = dst[e];
    const _Float16* qr = q + (size_t)d * QKV_DIM + h * HID;
    const _Float16* kr = k + (size_t)s * QKV_DIM + h * HID;
    float acc = 0.0f;
#pragma unroll
    for (int c = 0; c < HID; ++c) acc += (float)qr[c] * (float)kr[c];
    acc *= 0.17677669529663687f;              // 1/sqrt(32)
    alpha[idx] = acc;
    atomicMaxF(&mmax[d * HEADS + h], acc);
}

// p = exp(alpha - m[dst]); alpha <- p; segment-sum into ssum[dst,h]
__global__ void edge_pass_b(const int* __restrict__ dst, float* __restrict__ alpha,
                            const float* __restrict__ mmax, float* __restrict__ ssum, int E) {
    int idx = blockIdx.x * blockDim.x + threadIdx.x;
    if (idx >= E * HEADS) return;
    int e = idx >> 2, h = idx & 3;
    int d = dst[e];
    float p = __expf(alpha[idx] - mmax[d * HEADS + h]);
    alpha[idx] = p;
    atomicAdd(&ssum[d * HEADS + h], p);
}

// a = p / s[dst]; agg[dst] += a * v[src]
__global__ void edge_pass_c(const _Float16* __restrict__ v,
                            const int* __restrict__ src, const int* __restrict__ dst,
                            const float* __restrict__ alpha, const float* __restrict__ ssum,
                            float* __restrict__ agg, int E) {
    int idx = blockIdx.x * blockDim.x + threadIdx.x;
    if (idx >= E * HEADS) return;
    int e = idx >> 2, h = idx & 3;
    int s = src[e], d = dst[e];
    float a = alpha[idx] / ssum[d * HEADS + h];
    const _Float16* vr = v + (size_t)s * QKV_DIM + h * HID;
    float* ag = agg + (size_t)d * QKV_DIM + h * HID;
#pragma unroll
    for (int c = 0; c < HID; ++c) atomicAdd(&ag[c], a * (float)vr[c]);
}

// h = relu(mean_heads(agg) + skip), emitted as f16 for the next WMMA GEMM
__global__ void finalize_layer(const float* __restrict__ agg, const float* __restrict__ skip,
                               _Float16* __restrict__ h16, int n) {
    int idx = blockIdx.x * blockDim.x + threadIdx.x;
    if (idx >= n * HID) return;
    int node = idx / HID, c = idx - node * HID;
    const float* ar = agg + (size_t)node * QKV_DIM;
    float m = 0.25f * (ar[c] + ar[HID + c] + ar[2 * HID + c] + ar[3 * HID + c]) + skip[idx];
    h16[idx] = (_Float16)fmaxf(m, 0.0f);
}

// ---------------- driver ----------------

static inline int cdiv(int a, int b) { return (a + b - 1) / b; }

extern "C" void kernel_launch(void* const* d_in, const int* in_sizes, int n_in,
                              void* d_out, int out_size, void* d_ws, size_t ws_size,
                              hipStream_t stream) {
    const float* x    = (const float*)d_in[0];
    const int*   edge = (const int*)d_in[1];
    const int*   src  = edge;
    const int*   dst  = edge + N_EDGES;

    // per-layer weight pointers: Wq,bq,Wk,bk,Wv,bv,Ws,bs starting at index 2
    const float* W[2][4];  const float* B[2][4];
    for (int l = 0; l < 2; ++l)
        for (int j = 0; j < 4; ++j) {
            W[l][j] = (const float*)d_in[2 + l * 8 + 2 * j];
            B[l][j] = (const float*)d_in[2 + l * 8 + 2 * j + 1];
        }
    const float* Wout = (const float*)d_in[18];
    const float* bout = (const float*)d_in[19];

    // ---- carve workspace ----
    char* wp = (char*)d_ws;
    auto carve = [&](size_t bytes) { void* p = (void*)wp; wp += (bytes + 255) & ~(size_t)255; return p; };
    _Float16* xh16  = (_Float16*)carve((size_t)N_NODES * IN_DIM  * 2);
    _Float16* q16   = (_Float16*)carve((size_t)N_NODES * QKV_DIM * 2);
    _Float16* k16   = (_Float16*)carve((size_t)N_NODES * QKV_DIM * 2);
    _Float16* v16   = (_Float16*)carve((size_t)N_NODES * QKV_DIM * 2);
    float*    skip  = (float*)   carve((size_t)N_NODES * HID     * 4);
    float*    alpha = (float*)   carve((size_t)N_EDGES * HEADS   * 4);
    float*    mmax  = (float*)   carve((size_t)N_NODES * HEADS   * 4);
    float*    ssum  = (float*)   carve((size_t)N_NODES * HEADS   * 4);
    float*    agg   = (float*)   carve((size_t)N_NODES * QKV_DIM * 4);
    // f16 transposed weights
    _Float16* WT[2][4];
    int dins[2] = { IN_DIM, HID };
    for (int l = 0; l < 2; ++l) {
        WT[l][0] = (_Float16*)carve((size_t)QKV_DIM * dins[l] * 2);  // Wq^T
        WT[l][1] = (_Float16*)carve((size_t)QKV_DIM * dins[l] * 2);  // Wk^T
        WT[l][2] = (_Float16*)carve((size_t)QKV_DIM * dins[l] * 2);  // Wv^T
        WT[l][3] = (_Float16*)carve((size_t)HID     * dins[l] * 2);  // Ws^T
    }
    _Float16* WoutT = (_Float16*)carve((size_t)OUT_DIM * HID * 2);

    const int TPB = 256;
    const int GEMM_GRID = cdiv(N_NODES / 16, 4);   // 4 waves / 128-thread block

    // ---- weight prep ----
    for (int l = 0; l < 2; ++l) {
        int D = dins[l];
        for (int j = 0; j < 3; ++j)
            wt_prep<<<cdiv(D * QKV_DIM, TPB), TPB, 0, stream>>>(W[l][j], WT[l][j], D, QKV_DIM);
        wt_prep<<<cdiv(D * HID, TPB), TPB, 0, stream>>>(W[l][3], WT[l][3], D, HID);
    }
    wt_prep<<<cdiv(HID * OUT_DIM, TPB), TPB, 0, stream>>>(Wout, WoutT, HID, OUT_DIM);

    // ---- layer input: x -> f16 ----
    cvt_f32_f16<<<cdiv(N_NODES * IN_DIM, TPB), TPB, 0, stream>>>(x, xh16, N_NODES * IN_DIM);

    const float NEG_INF = -__builtin_huge_valf();

    for (int l = 0; l < 2; ++l) {
        // projections: q,k,v (f16 out, F=128 -> NT=8) and skip (f32 out, F=32 -> NT=2)
        if (l == 0) {
            wmma_gemm<IN_DIM, 8, 1><<<GEMM_GRID, 128, 0, stream>>>(xh16, WT[0][0], B[0][0], q16, N_NODES);
            wmma_gemm<IN_DIM, 8, 1><<<GEMM_GRID, 128, 0, stream>>>(xh16, WT[0][1], B[0][1], k16, N_NODES);
            wmma_gemm<IN_DIM, 8, 1><<<GEMM_GRID, 128, 0, stream>>>(xh16, WT[0][2], B[0][2], v16, N_NODES);
            wmma_gemm<IN_DIM, 2, 0><<<GEMM_GRID, 128, 0, stream>>>(xh16, WT[0][3], B[0][3], skip, N_NODES);
        } else {
            wmma_gemm<HID, 8, 1><<<GEMM_GRID, 128, 0, stream>>>(xh16, WT[1][0], B[1][0], q16, N_NODES);
            wmma_gemm<HID, 8, 1><<<GEMM_GRID, 128, 0, stream>>>(xh16, WT[1][1], B[1][1], k16, N_NODES);
            wmma_gemm<HID, 8, 1><<<GEMM_GRID, 128, 0, stream>>>(xh16, WT[1][2], B[1][2], v16, N_NODES);
            wmma_gemm<HID, 2, 0><<<GEMM_GRID, 128, 0, stream>>>(xh16, WT[1][3], B[1][3], skip, N_NODES);
        }

        // reset segment buffers
        fill_f32<<<cdiv(N_NODES * HEADS, TPB), TPB, 0, stream>>>(mmax, NEG_INF, N_NODES * HEADS);
        fill_f32<<<cdiv(N_NODES * HEADS, TPB), TPB, 0, stream>>>(ssum, 0.0f, N_NODES * HEADS);
        fill_f32<<<cdiv(N_NODES * QKV_DIM, TPB), TPB, 0, stream>>>(agg, 0.0f, N_NODES * QKV_DIM);

        // softmax-aggregate over edges (q/k/v are L2-resident: 25.6 MB each vs 192 MB L2)
        int eg = cdiv(N_EDGES * HEADS, TPB);
        edge_pass_a<<<eg, TPB, 0, stream>>>(q16, k16, src, dst, alpha, mmax, N_EDGES);
        edge_pass_b<<<eg, TPB, 0, stream>>>(dst, alpha, mmax, ssum, N_EDGES);
        edge_pass_c<<<eg, TPB, 0, stream>>>(v16, src, dst, alpha, ssum, agg, N_EDGES);

        // h = relu(mean_heads(agg) + skip) -> f16 into xh16 (next layer input, D=32)
        finalize_layer<<<cdiv(N_NODES * HID, TPB), TPB, 0, stream>>>(agg, skip, xh16, N_NODES);
    }

    // output head: d_out = h @ Wout + bout  (f32 out, F=16 -> NT=1)
    wmma_gemm<HID, 1, 0><<<GEMM_GRID, 128, 0, stream>>>(xh16, WoutT, bout, d_out, N_NODES);
}
